// Attention_50216757624885
// MI455X (gfx1250) — matrix operand order repaired and checked by reading, not compiled, not run on hardware
//
#include <hip/hip_runtime.h>

#define S_LEN   2048
#define D_MODEL 4096
#define N_HEADS 32
#define N_KV    8
#define HEAD_DIM 128
#define QKV_N   6144   /* 4096 q + 1024 k + 1024 v */
#define NEGV    (-1.0e9f)
#define SCALE_QK 0.08838834764831845f  /* 1/sqrt(128) */

typedef __attribute__((ext_vector_type(16))) __bf16        v16bf;
typedef __attribute__((ext_vector_type(8)))  float         v8f;
typedef __attribute__((ext_vector_type(4)))  float         f32x4;
typedef __attribute__((ext_vector_type(2)))  unsigned int  u32x2;
typedef __attribute__((ext_vector_type(4)))  unsigned int  u32x4;
typedef __attribute__((ext_vector_type(8)))  unsigned int  u32x8;
typedef __attribute__((ext_vector_type(4)))  int           i32x4;
typedef __attribute__((ext_vector_type(8)))  int           i32x8;

#ifndef __has_builtin
#define __has_builtin(x) 0
#endif
#if __has_builtin(__builtin_amdgcn_tensor_load_to_lds)
#define HAVE_TDM 1
#else
#define HAVE_TDM 0   /* host pass / toolchains without the builtin: sync staging */
#endif

// ---------------- bf16 helpers (bit-exact RNE)
__device__ __forceinline__ unsigned short f2bfbits(float f) {
  unsigned int u = __builtin_bit_cast(unsigned int, f);
  u += 0x7FFFu + ((u >> 16) & 1u);
  return (unsigned short)(u >> 16);
}

// ---------------- fragment builders: two contiguous 16B runs -> v16bf
__device__ __forceinline__ v16bf frag_from2(const unsigned short* p0, const unsigned short* p1) {
  u32x4 a = *(const u32x4*)p0;
  u32x4 b = *(const u32x4*)p1;
  u32x8 u;
  u[0] = a[0]; u[1] = a[1]; u[2] = a[2]; u[3] = a[3];
  u[4] = b[0]; u[5] = b[1]; u[6] = b[2]; u[7] = b[3];
  return __builtin_bit_cast(v16bf, u);
}
// A 16x32 bf16 (ISA 7.12.2): lane m=lane&15, half=lane>>4; chunks k0+half*8, k0+16+half*8
__device__ __forceinline__ v16bf loadA_bf16v(const unsigned short* A, int lda, int m0, int k0, int lane) {
  const unsigned short* base =
      A + (size_t)(m0 + (lane & 15)) * lda + k0 + ((lane >> 4) << 3);
  return frag_from2(base, base + 16);
}
// B 32x16 bf16 from k-contiguous rows: lane n=lane&15, half; chunks k0+half*8, k0+16+half*8
__device__ __forceinline__ v16bf loadB_bf16v(const unsigned short* Bp, int ldb, int n0, int k0, int lane) {
  const unsigned short* base =
      Bp + (size_t)(n0 + (lane & 15)) * ldb + k0 + ((lane >> 4) << 3);
  return frag_from2(base, base + 16);
}

// ---------------- TDM: 2D tile (32 k-elems x 64 rows, bf16) global -> LDS (ISA ch.8 D#)
__device__ __forceinline__ void tdm_issue_b_tile(const void* gptr, unsigned int ldsOff,
                                                 unsigned int kElems, unsigned int nRows) {
#if HAVE_TDM
  const unsigned long long ga = (unsigned long long)gptr;
  u32x4 g0;
  g0[0] = 1u;                                   // count=1, user mode, no gather
  g0[1] = ldsOff;                               // lds_addr (bytes)
  g0[2] = (unsigned int)ga;                     // global_addr[31:0]
  g0[3] = (unsigned int)((ga >> 32) & 0x01FFFFFFu) | (2u << 30);  // addr[56:32] | type=2
  u32x8 g1u;
  g1u[0] = (1u << 16);                          // data_size=1 (2 bytes)
  g1u[1] = (kElems & 0xFFFFu) << 16;            // tensor_dim0[15:0]
  g1u[2] = (kElems >> 16) | ((nRows & 0xFFFFu) << 16);  // dim0[31:16] | dim1[15:0]
  g1u[3] = (nRows >> 16) | (32u << 16);         // dim1[31:16] | tile_dim0=32
  g1u[4] = 64u;                                 // tile_dim1=64, tile_dim2=0
  g1u[5] = kElems;                              // tensor_dim0_stride[31:0]
  g1u[6] = 0u;                                  // stride0[47:32]=0 | stride1[15:0]=0
  g1u[7] = 0u;
  i32x8 g1 = __builtin_bit_cast(i32x8, g1u);
  i32x4 z4 = {0, 0, 0, 0};
#if __clang_major__ >= 23
  i32x8 z8 = {0, 0, 0, 0, 0, 0, 0, 0};
  __builtin_amdgcn_tensor_load_to_lds(g0, g1, z4, z4, z8, 0);
#else
  __builtin_amdgcn_tensor_load_to_lds(g0, g1, z4, z4, 0);
#endif
#else
  (void)gptr; (void)ldsOff; (void)kElems; (void)nRows;
#endif
}

// ---------------- wave-level LDS ordering (attention waves diverge -> no __syncthreads)
__device__ __forceinline__ void lds_fence_wave() {
  __builtin_amdgcn_wave_barrier();
  asm volatile("s_wait_dscnt 0" ::: "memory");
  __builtin_amdgcn_wave_barrier();
}

// ---------------- 16-lane (half-wave) reductions across C-fragment column lanes
__device__ __forceinline__ float rmax16(float v) {
  v = fmaxf(v, __shfl_xor(v, 1, 32));
  v = fmaxf(v, __shfl_xor(v, 2, 32));
  v = fmaxf(v, __shfl_xor(v, 4, 32));
  v = fmaxf(v, __shfl_xor(v, 8, 32));
  return v;
}
__device__ __forceinline__ float rsum16(float v) {
  v += __shfl_xor(v, 1, 32);
  v += __shfl_xor(v, 2, 32);
  v += __shfl_xor(v, 4, 32);
  v += __shfl_xor(v, 8, 32);
  return v;
}

// ---------------- one-time fp32 -> bf16 conversion (vectorized 4-wide)
__global__ void f32_to_bf16_kernel(const float* __restrict__ src,
                                   unsigned short* __restrict__ dst, long long n4) {
  long long i = (long long)blockIdx.x * blockDim.x + threadIdx.x;
  if (i >= n4) return;
  f32x4 f = *(const f32x4*)(src + 4 * i);
  u32x2 o;
  o[0] = ((unsigned int)f2bfbits(f[1]) << 16) | f2bfbits(f[0]);
  o[1] = ((unsigned int)f2bfbits(f[3]) << 16) | f2bfbits(f[2]);
  *(u32x2*)(dst + 4 * i) = o;
}

// ---------------- GEMM: C(M,N) = A(M,K) * W(N,K)^T, bf16 WMMA, fp32 accumulate
// block: 256 thr = 8 waves -> 256(M) x 64(N); wave tile 32x64 (8 WMMA / k-step).
// B tile (64x32 bf16, 4KB) double-buffered in LDS via TDM (async DMA overlapped with
// WMMA burst, retired with s_wait_tensorcnt), sync-staging fallback otherwise.
__global__ void __launch_bounds__(256)
gemm_bf16wmma_kernel(const unsigned short* __restrict__ A,
                     const unsigned short* __restrict__ W,
                     float* __restrict__ C, int M, int N, int K) {
  __shared__ unsigned short bTile[2][64 * 32];
  const int tid  = threadIdx.x;
  const int lane = tid & 31, wave = tid >> 5;
  const int m0 = blockIdx.y * 256 + wave * 32;
  const int n0 = blockIdx.x * 64;
  const int n = lane & 15, half = lane >> 4;

  const unsigned short* wt = W + (size_t)n0 * K;  // B tile column origin

#if HAVE_TDM
  const unsigned int lds0 = (unsigned int)(unsigned long long)(&bTile[0][0]);
  if (wave == 0) {
    tdm_issue_b_tile(wt, lds0, (unsigned)K, (unsigned)N);
    __builtin_amdgcn_s_wait_tensorcnt(0);
  }
  __syncthreads();
#else
  const int lrow = tid >> 2;         // 0..63 : B row within tile
  const int lseg = (tid & 3) << 3;   // 0,8,16,24 : 16B segment
  const unsigned short* wsrc = wt + (size_t)lrow * K + lseg;
  *(u32x4*)&bTile[0][lrow * 32 + lseg] = *(const u32x4*)wsrc;
  __syncthreads();
#endif

  v8f acc[8] = {};
  for (int k0 = 0; k0 < K; k0 += 32) {
    const int  cur     = (k0 >> 5) & 1;
    const bool hasNext = (k0 + 32) < K;
#if HAVE_TDM
    if (hasNext && wave == 0)
      tdm_issue_b_tile(wt + k0 + 32, lds0 + (unsigned)((cur ^ 1) * 4096),
                       (unsigned)K, (unsigned)N);
#else
    u32x4 wnext;
    if (hasNext) {
      wnext = *(const u32x4*)(wsrc + k0 + 32);
      __builtin_prefetch(wsrc + k0 + 64, 0, 1);  // -> global_prefetch_b8
    }
#endif
    v16bf a0 = loadA_bf16v(A, K, m0,      k0, lane);
    v16bf a1 = loadA_bf16v(A, K, m0 + 16, k0, lane);
    v16bf bf[4];
#pragma unroll
    for (int t = 0; t < 4; ++t) {
      const unsigned short* bb = &bTile[cur][(t * 16 + n) * 32 + (half << 3)];
      bf[t] = frag_from2(bb, bb + 16);
    }
#pragma unroll
    for (int t = 0; t < 4; ++t) {
      acc[t]     = __builtin_amdgcn_wmma_f32_16x16x32_bf16(false, a0, false, bf[t],
                                                           (short)0, acc[t], false, false);
      acc[4 + t] = __builtin_amdgcn_wmma_f32_16x16x32_bf16(false, a1, false, bf[t],
                                                           (short)0, acc[4 + t], false, false);
    }
#if HAVE_TDM
    if (wave == 0) __builtin_amdgcn_s_wait_tensorcnt(0);
#else
    if (hasNext) *(u32x4*)&bTile[cur ^ 1][lrow * 32 + lseg] = wnext;
#endif
    __syncthreads();
  }

#pragma unroll
  for (int ms = 0; ms < 2; ++ms)
#pragma unroll
    for (int t = 0; t < 4; ++t)
#pragma unroll
      for (int r = 0; r < 8; ++r)
        C[(size_t)(m0 + ms * 16 + (half << 3) + r) * N + (n0 + t * 16 + n)] =
            acc[ms * 4 + t][r];
}

// ---------------- RoPE (q,k) + bf16 repack; V stored TRANSPOSED [kv][d][s]
__global__ void rope_pack_kernel(const float* __restrict__ Y,
                                 const float* __restrict__ cosT,
                                 const float* __restrict__ sinT,
                                 unsigned short* __restrict__ Qb,
                                 unsigned short* __restrict__ Kb,
                                 unsigned short* __restrict__ Vt) {
  const long long QP = (long long)S_LEN * 2048;
  const long long KP = (long long)S_LEN * 512;
  const long long VN = (long long)S_LEN * 1024;
  long long id = (long long)blockIdx.x * blockDim.x + threadIdx.x;
  if (id < QP) {
    const int s = (int)(id >> 11), p = (int)(id & 2047);
    const int h = p >> 6, i = p & 63;
    const float xr = Y[(size_t)s * QKV_N + h * 128 + 2 * i];
    const float xi = Y[(size_t)s * QKV_N + h * 128 + 2 * i + 1];
    const float c = cosT[s * 64 + i], sn = sinT[s * 64 + i];
    const size_t o = (size_t)h * S_LEN * HEAD_DIM + (size_t)s * HEAD_DIM + 2 * i;
    Qb[o]     = f2bfbits(xr * c - xi * sn);
    Qb[o + 1] = f2bfbits(xr * sn + xi * c);
  } else if (id < QP + KP) {
    const long long t = id - QP;
    const int s = (int)(t >> 9), p = (int)(t & 511);
    const int kh = p >> 6, i = p & 63;
    const float xr = Y[(size_t)s * QKV_N + 4096 + kh * 128 + 2 * i];
    const float xi = Y[(size_t)s * QKV_N + 4096 + kh * 128 + 2 * i + 1];
    const float c = cosT[s * 64 + i], sn = sinT[s * 64 + i];
    const size_t o = (size_t)kh * S_LEN * HEAD_DIM + (size_t)s * HEAD_DIM + 2 * i;
    Kb[o]     = f2bfbits(xr * c - xi * sn);
    Kb[o + 1] = f2bfbits(xr * sn + xi * c);
  } else if (id < QP + KP + VN) {
    const long long t = id - QP - KP;
    const int s = (int)(t >> 10), p = (int)(t & 1023);
    Vt[(size_t)(p >> 7) * HEAD_DIM * S_LEN + (size_t)(p & 127) * S_LEN + s] =
        f2bfbits(Y[(size_t)s * QKV_N + 5120 + p]);
  }
}

// ---------------- flash attention: one wave = (head h, 16-row q-tile), causal, GQA
__global__ void __launch_bounds__(256)
attn_kernel(const unsigned short* __restrict__ Qb,
            const unsigned short* __restrict__ Kb,
            const unsigned short* __restrict__ Vt,
            unsigned short* __restrict__ Ob) {
  __shared__ unsigned short pShm[8 * 512];  // per-wave 16x32 bf16 P staging
  const int lane = threadIdx.x & 31;
  const int wave = threadIdx.x >> 5;
  const int gw = blockIdx.x * 8 + wave;
  const int h  = gw >> 7;
  const int qt = gw & 127;
  const int q0 = qt << 4;
  const int kvh = h >> 2;
  const unsigned short* Qh = Qb + (size_t)h   * S_LEN * HEAD_DIM;
  const unsigned short* Kh = Kb + (size_t)kvh * S_LEN * HEAD_DIM;
  const unsigned short* Vh = Vt + (size_t)kvh * HEAD_DIM * S_LEN;  // [d][s]
  unsigned short* pLds = &pShm[wave * 512];

  const int n = lane & 15, half = lane >> 4;

  v16bf qf[4];
#pragma unroll
  for (int j = 0; j < 4; ++j) qf[j] = loadA_bf16v(Qh, HEAD_DIM, q0, j * 32, lane);

  v8f acc[8] = {};
  float mrow[8], lrow[8];
#pragma unroll
  for (int r = 0; r < 8; ++r) { mrow[r] = -3.0e38f; lrow[r] = 0.0f; }

  const int kend = q0 + 16;
  for (int kb = 0; kb < kend; kb += 32) {
    const bool have1 = (kb + 16 < kend);

    v8f s0 = {}, s1 = {};
#pragma unroll
    for (int j = 0; j < 4; ++j) {
      v16bf b0 = loadB_bf16v(Kh, HEAD_DIM, kb, j * 32, lane);
      s0 = __builtin_amdgcn_wmma_f32_16x16x32_bf16(false, qf[j], false, b0,
                                                   (short)0, s0, false, false);
    }
    if (have1) {
#pragma unroll
      for (int j = 0; j < 4; ++j) {
        v16bf b1 = loadB_bf16v(Kh, HEAD_DIM, kb + 16, j * 32, lane);
        s1 = __builtin_amdgcn_wmma_f32_16x16x32_bf16(false, qf[j], false, b1,
                                                     (short)0, s1, false, false);
      }
    }

#pragma unroll
    for (int r = 0; r < 8; ++r) {
      const int m = q0 + (half << 3) + r;
      float v0 = s0[r] * SCALE_QK + ((kb + n) <= m ? 0.0f : NEGV);
      float v1 = have1 ? (s1[r] * SCALE_QK + ((kb + 16 + n) <= m ? 0.0f : NEGV)) : NEGV;
      float tmax = rmax16(fmaxf(v0, v1));
      const float mnew = fmaxf(mrow[r], tmax);
      const float so = __expf(mrow[r] - mnew);
      const float e0 = __expf(v0 - mnew);
      const float e1 = __expf(v1 - mnew);
      lrow[r] = lrow[r] * so + rsum16(e0 + e1);
      mrow[r] = mnew;
      const int row = (half << 3) + r;
      pLds[row * 32 + n]      = f2bfbits(e0);
      pLds[row * 32 + 16 + n] = f2bfbits(e1);
#pragma unroll
      for (int t = 0; t < 8; ++t) acc[t][r] *= so;
    }

    lds_fence_wave();
    const unsigned short* pbase = &pLds[(lane & 15) * 32 + ((lane >> 4) << 3)];
    v16bf pf = frag_from2(pbase, pbase + 16);
    lds_fence_wave();

#pragma unroll
    for (int t = 0; t < 8; ++t) {
      v16bf vb = loadB_bf16v(Vh, S_LEN, t * 16, kb, lane);
      acc[t] = __builtin_amdgcn_wmma_f32_16x16x32_bf16(false, pf, false, vb,
                                                       (short)0, acc[t], false, false);
    }
  }

#pragma unroll
  for (int r = 0; r < 8; ++r) {
    const float inv = 1.0f / lrow[r];
    const int row = q0 + (half << 3) + r;
#pragma unroll
    for (int t = 0; t < 8; ++t)
      Ob[(size_t)row * (N_HEADS * HEAD_DIM) + h * HEAD_DIM + t * 16 + n] =
          f2bfbits(acc[t][r] * inv);
  }
}

extern "C" void kernel_launch(void* const* d_in, const int* in_sizes, int n_in,
                              void* d_out, int out_size, void* d_ws, size_t ws_size,
                              hipStream_t stream) {
  (void)in_sizes; (void)n_in; (void)out_size; (void)ws_size;
  const float* x    = (const float*)d_in[0];
  const float* wq   = (const float*)d_in[1];
  const float* wk   = (const float*)d_in[2];
  const float* wv   = (const float*)d_in[3];
  const float* wo   = (const float*)d_in[4];
  const float* cosT = (const float*)d_in[7];
  const float* sinT = (const float*)d_in[8];
  float* out = (float*)d_out;

  // workspace layout (~193 MB)
  float*          Y    = (float*)d_ws;                                   // (S,6144) f32
  unsigned short* Xb   = (unsigned short*)(Y + (size_t)S_LEN * QKV_N);   // (S,4096)
  unsigned short* Wqkv = Xb + (size_t)S_LEN * D_MODEL;                   // (6144,4096)
  unsigned short* Wob  = Wqkv + (size_t)QKV_N * D_MODEL;                 // (4096,4096)
  unsigned short* Qb   = Wob + (size_t)D_MODEL * D_MODEL;                // (H,S,HD)
  unsigned short* Kb   = Qb + (size_t)S_LEN * D_MODEL;                   // (KV,S,HD)
  unsigned short* Vt   = Kb + (size_t)S_LEN * (N_KV * HEAD_DIM);         // (KV,HD,S)
  unsigned short* Ob   = Vt + (size_t)S_LEN * (N_KV * HEAD_DIM);         // (S,H*HD)

  auto cvt = [&](const float* s, unsigned short* d, long long nelem) {
    long long n4 = nelem >> 2;
    f32_to_bf16_kernel<<<(int)((n4 + 255) / 256), 256, 0, stream>>>(s, d, n4);
  };
  cvt(x,  Xb, (long long)S_LEN * D_MODEL);
  cvt(wq, Wqkv,                          (long long)4096 * D_MODEL);
  cvt(wk, Wqkv + (size_t)4096 * D_MODEL, (long long)1024 * D_MODEL);
  cvt(wv, Wqkv + (size_t)5120 * D_MODEL, (long long)1024 * D_MODEL);
  cvt(wo, Wob, (long long)D_MODEL * D_MODEL);

  // 1) fused QKV projection: Y = x * [wq|wk|wv]^T
  gemm_bf16wmma_kernel<<<dim3(QKV_N / 64, S_LEN / 256), 256, 0, stream>>>(
      Xb, Wqkv, Y, S_LEN, QKV_N, D_MODEL);

  // 2) RoPE + bf16 repack (V transposed)
  const long long total = (long long)S_LEN * (2048 + 512 + 1024);
  rope_pack_kernel<<<(int)((total + 255) / 256), 256, 0, stream>>>(
      Y, cosT, sinT, Qb, Kb, Vt);

  // 3) causal flash attention (GQA)
  attn_kernel<<<(N_HEADS * (S_LEN / 16)) / 8, 256, 0, stream>>>(Qb, Kb, Vt, Ob);

  // 4) output projection: out = O * wo^T
  gemm_bf16wmma_kernel<<<dim3(D_MODEL / 64, S_LEN / 256), 256, 0, stream>>>(
      Ob, Wob, out, S_LEN, D_MODEL, D_MODEL);
}
